// yueGPT_79035988181353
// MI455X (gfx1250) — compile-verified, matrix operand order checked by reading, not run on hardware
//
#include <hip/hip_runtime.h>
#include <hip/hip_bf16.h>
#include <math.h>

#define V_SZ   32000
#define E_SZ   512
#define H_SZ   8
#define L_SZ   6
#define HS_SZ  64
#define B_SZ   64
#define T_SZ   64
#define M_TOK  (B_SZ * T_SZ)   // 4096 tokens
#define FF_SZ  (4 * E_SZ)      // 2048
#define LN_EPS 1e-5f

typedef __attribute__((ext_vector_type(16))) __bf16 v16bf;
typedef __attribute__((ext_vector_type(8)))  float  v8f;

// f32 <-> bf16 helpers (pure bit ops, RNE)
__device__ __forceinline__ __bf16 f2bf(float f) {
  unsigned u = __builtin_bit_cast(unsigned, f);
  unsigned r = (u + 0x7FFFu + ((u >> 16) & 1u)) >> 16;
  return __builtin_bit_cast(__bf16, (unsigned short)r);
}
__device__ __forceinline__ float bf2f(__bf16 b) {
  unsigned short u = __builtin_bit_cast(unsigned short, b);
  return __builtin_bit_cast(float, (unsigned)u << 16);
}

// ------------------------------------------------------- weight swizzle
// f32 weight (K,N) -> bf16 in WMMA B-fragment order:
// out[((nt*(K/32) + kt)*512) + lane*16 + i] = W[k][n]
//   n = nt*16 + (lane&15), k = kt*32 + (i>=8)*16 + (lane>>4)*8 + (i&7)
// wkmode: W is Wk layout (H,E,HS): W[k][n] = Wk[(n>>6)*K*64 + k*64 + (n&63)]
__global__ __launch_bounds__(32) void wswz_kernel(const float* __restrict__ W,
                                                  __bf16* __restrict__ out,
                                                  int K, int N, int wkmode) {
  int nt = blockIdx.x, kt = blockIdx.y;
  int lane = threadIdx.x, lh = lane >> 4, lm = lane & 15;
  int n = nt * 16 + lm;
  long base = ((long)nt * (K >> 5) + kt) * 512 + (long)lane * 16;
#pragma unroll
  for (int i = 0; i < 16; i++) {
    int k = kt * 32 + ((i >> 3) << 4) + lh * 8 + (i & 7);
    long off = wkmode ? ((long)(n >> 6) * K * 64 + (long)k * 64 + (n & 63))
                      : ((long)k * N + n);
    out[base + i] = f2bf(W[off]);
  }
}

// ------------------------------------------------------- f32 -> bf16 copy
__global__ void cvt_kernel(const float* __restrict__ in, __bf16* __restrict__ out, long n) {
  long i = (long)blockIdx.x * blockDim.x + threadIdx.x;
  if (i < n) out[i] = f2bf(in[i]);
}

// ---------------------------------------------------------------- embedding
__global__ void embed_kernel(const int* __restrict__ inputs,
                             const float* __restrict__ emb,
                             const float* __restrict__ pos,
                             float* __restrict__ x) {
  int tok = blockIdx.x;            // b*T + t
  int t   = tok & (T_SZ - 1);
  long idx = inputs[tok];
  const float* er = emb + idx * (long)E_SZ;
  const float* pr = pos + (long)t * E_SZ;
  float* xr = x + (long)tok * E_SZ;
  for (int e = threadIdx.x; e < E_SZ; e += blockDim.x)
    xr[e] = er[e] + pr[e];
}

// ---------------------------------------------------------------- layernorm
// one wave32 per token, 8 tokens per 256-thread block; bf16 output
__global__ __launch_bounds__(256) void ln_kernel(const float* __restrict__ x,
                                                 const float* __restrict__ g,
                                                 const float* __restrict__ b,
                                                 __bf16* __restrict__ h) {
  int wave = threadIdx.x >> 5;
  int lane = threadIdx.x & 31;
  int tok  = blockIdx.x * 8 + wave;
  const float* xr = x + (long)tok * E_SZ;
  float v[16];
  float s = 0.f;
#pragma unroll
  for (int i = 0; i < 16; i++) { v[i] = xr[lane + i * 32]; s += v[i]; }
#pragma unroll
  for (int o = 16; o > 0; o >>= 1) s += __shfl_xor(s, o, 32);
  float mean = s * (1.0f / E_SZ);
  float q = 0.f;
#pragma unroll
  for (int i = 0; i < 16; i++) { float d = v[i] - mean; q += d * d; }
#pragma unroll
  for (int o = 16; o > 0; o >>= 1) q += __shfl_xor(q, o, 32);
  float inv = rsqrtf(q * (1.0f / E_SZ) + LN_EPS);
  __bf16* hr = h + (long)tok * E_SZ;
#pragma unroll
  for (int i = 0; i < 16; i++) {
    int e = lane + i * 32;
    hr[e] = f2bf((v[i] - mean) * inv * g[e] + b[e]);
  }
}

// ---------------------------------------------------------------- WMMA GEMM
// C[M,N] = epilogue( A[M,K](bf16) x Bsw[K,N](bf16,swizzled) + bias [+ Res] )
// block = 256 thr (8 waves), tile 256x64; wave -> 32 rows x 64 cols (2x4 accs)
// per k-step: issue all 12 b128 loads (distinct regs), then 8 WMMAs in
// operand-arrival order so the compiler can use partial s_wait_loadcnt.
#define GF_RELU  1
#define GF_RES   2
#define GF_OUTBF 4

#define WMMA_BF16(A_, B_, C_) __builtin_amdgcn_wmma_f32_16x16x32_bf16( \
    false, (A_), false, (B_), (short)0, (C_), false, false)

__global__ __launch_bounds__(256) void gemm_kernel(
    const __bf16* __restrict__ A, const __bf16* __restrict__ Bsw,
    const float* __restrict__ bias, const float* __restrict__ Res,
    void* __restrict__ Cout, int M, int N, int K, int flags) {
  int wave = threadIdx.x >> 5;
  int lane = threadIdx.x & 31;
  int lh   = lane >> 4;
  int lm   = lane & 15;
  int m0   = blockIdx.y * 256 + wave * 32;
  int n0   = blockIdx.x * 64;
  int ksn  = K >> 5;                 // k-steps

  v8f acc[2][4] = {};

  const __bf16* arow0 = A + (long)(m0 + lm) * K + lh * 8;
  const __bf16* arow1 = arow0 + (long)16 * K;
  const __bf16* bbase = Bsw + (long)(blockIdx.x * 4) * ksn * 512 + (long)lane * 16;
  const long bstride = (long)ksn * 512;   // elements between j-fragments

  for (int ks = 0; ks < ksn; ks++) {
    const __bf16* ap0 = arow0 + ks * 32;
    const __bf16* ap1 = arow1 + ks * 32;
    const __bf16* bp0 = bbase + (long)ks * 512;
    const __bf16* bp1 = bp0 + bstride;
    const __bf16* bp2 = bp1 + bstride;
    const __bf16* bp3 = bp2 + bstride;

    v16bf a0, a1, b0, b1, b2, b3;
    ((uint4*)&a0)[0] = *(const uint4*)(ap0);
    ((uint4*)&a0)[1] = *(const uint4*)(ap0 + 16);
    ((uint4*)&a1)[0] = *(const uint4*)(ap1);
    ((uint4*)&a1)[1] = *(const uint4*)(ap1 + 16);
    ((uint4*)&b0)[0] = *(const uint4*)(bp0);
    ((uint4*)&b0)[1] = *(const uint4*)(bp0 + 8);
    ((uint4*)&b1)[0] = *(const uint4*)(bp1);
    ((uint4*)&b1)[1] = *(const uint4*)(bp1 + 8);
    ((uint4*)&b2)[0] = *(const uint4*)(bp2);
    ((uint4*)&b2)[1] = *(const uint4*)(bp2 + 8);
    ((uint4*)&b3)[0] = *(const uint4*)(bp3);
    ((uint4*)&b3)[1] = *(const uint4*)(bp3 + 8);

    acc[0][0] = WMMA_BF16(a0, b0, acc[0][0]);
    acc[1][0] = WMMA_BF16(a1, b0, acc[1][0]);
    acc[0][1] = WMMA_BF16(a0, b1, acc[0][1]);
    acc[1][1] = WMMA_BF16(a1, b1, acc[1][1]);
    acc[0][2] = WMMA_BF16(a0, b2, acc[0][2]);
    acc[1][2] = WMMA_BF16(a1, b2, acc[1][2]);
    acc[0][3] = WMMA_BF16(a0, b3, acc[0][3]);
    acc[1][3] = WMMA_BF16(a1, b3, acc[1][3]);
  }

  float*  Cf = (float*)Cout;
  __bf16* Cb = (__bf16*)Cout;
#pragma unroll
  for (int g = 0; g < 2; g++) {
#pragma unroll
    for (int j = 0; j < 4; j++) {
      int n = n0 + j * 16 + lm;
      float bb = bias ? bias[n] : 0.0f;
#pragma unroll
      for (int r = 0; r < 8; r++) {
        int m = m0 + g * 16 + lh * 8 + r;
        float vv = acc[g][j][r] + bb;
        if (flags & GF_RES)  vv += Res[(long)m * N + n];
        if (flags & GF_RELU) vv = fmaxf(vv, 0.0f);
        if (flags & GF_OUTBF) Cb[(long)m * N + n] = f2bf(vv);
        else                  Cf[(long)m * N + n] = vv;
      }
    }
  }
}

// ---------------------------------------------------------------- attention
// one block of 64 threads per (b,h); k==q==v per reference; scale = E^-0.5
__global__ __launch_bounds__(64) void attn_kernel(const __bf16* __restrict__ k,
                                                  __bf16* __restrict__ o) {
  __shared__ float kbT[HS_SZ][T_SZ + 1];   // transposed K tile, padded
  __shared__ float sc[T_SZ][T_SZ + 1];     // scores/probs, padded
  int bh = blockIdx.x;
  int b  = bh / H_SZ, h = bh % H_SZ;
  int t  = threadIdx.x;                    // 0..63, owns query row t
  const __bf16* kr = k + (long)(b * T_SZ + t) * E_SZ + h * HS_SZ;
  for (int s = 0; s < HS_SZ; s++) kbT[s][t] = bf2f(kr[s]);
  __syncthreads();

  const float scale = 0.044194173824159216f;  // 512^-0.5
  float mx = -INFINITY;
  for (int u = 0; u <= t; u++) {
    float d = 0.f;
#pragma unroll 8
    for (int s = 0; s < HS_SZ; s++) d += kbT[s][t] * kbT[s][u];
    d *= scale;
    sc[t][u] = d;
    mx = fmaxf(mx, d);
  }
  float den = 0.f;
  for (int u = 0; u <= t; u++) {
    float w = __expf(sc[t][u] - mx);
    sc[t][u] = w;
    den += w;
  }
  float inv = 1.0f / den;

  __bf16* orow = o + (long)(b * T_SZ + t) * E_SZ + h * HS_SZ;
  for (int s = 0; s < HS_SZ; s++) {
    float acc = 0.f;
    for (int u = 0; u <= t; u++) acc += sc[t][u] * kbT[s][u];
    orow[s] = f2bf(acc * inv);
  }
}

// ---------------------------------------------------------------- loss
__global__ void zero_loss_kernel(float* loss) {
  if (threadIdx.x == 0) loss[0] = 0.f;
}

__global__ __launch_bounds__(256) void loss_kernel(const float* __restrict__ logits,
                                                   const int* __restrict__ targets,
                                                   float* __restrict__ loss) {
  __shared__ float red[256];
  int r = blockIdx.x;
  const float* lr = logits + (long)r * V_SZ;
  float mx = -INFINITY;
  for (int i = threadIdx.x; i < V_SZ; i += 256) mx = fmaxf(mx, lr[i]);
  red[threadIdx.x] = mx; __syncthreads();
  for (int s = 128; s > 0; s >>= 1) {
    if (threadIdx.x < s) red[threadIdx.x] = fmaxf(red[threadIdx.x], red[threadIdx.x + s]);
    __syncthreads();
  }
  mx = red[0]; __syncthreads();
  float sum = 0.f;
  for (int i = threadIdx.x; i < V_SZ; i += 256) sum += __expf(lr[i] - mx);
  red[threadIdx.x] = sum; __syncthreads();
  for (int s = 128; s > 0; s >>= 1) {
    if (threadIdx.x < s) red[threadIdx.x] += red[threadIdx.x + s];
    __syncthreads();
  }
  if (threadIdx.x == 0) {
    float lse = mx + __logf(red[0]);
    float lp  = lr[targets[r]] - lse;
    atomicAdd(loss, -lp * (1.0f / M_TOK));
  }
}

// ---------------------------------------------------------------- launch
extern "C" void kernel_launch(void* const* d_in, const int* in_sizes, int n_in,
                              void* d_out, int out_size, void* d_ws, size_t ws_size,
                              hipStream_t stream) {
  const int*   inputs  = (const int*)  d_in[0];
  const int*   targets = (const int*)  d_in[1];
  const float* emb     = (const float*)d_in[2];
  const float* pos     = (const float*)d_in[3];
  const float* ln1_g   = (const float*)d_in[4];
  const float* ln1_b   = (const float*)d_in[5];
  const float* Wk      = (const float*)d_in[6];
  const float* Wproj   = (const float*)d_in[7];
  const float* bproj   = (const float*)d_in[8];
  const float* ln2_g   = (const float*)d_in[9];
  const float* ln2_b   = (const float*)d_in[10];
  const float* W1      = (const float*)d_in[11];
  const float* b1      = (const float*)d_in[12];
  const float* W2      = (const float*)d_in[13];
  const float* b2      = (const float*)d_in[14];
  const float* lm_w    = (const float*)d_in[15];
  const float* lm_b    = (const float*)d_in[16];

  float* logits = (float*)d_out;
  float* loss   = logits + (long)M_TOK * V_SZ;

  // workspace layout (bytes)
  char* ws = (char*)d_ws;
  long off = 0;
  float*  x   = (float*) (ws + off); off += (long)M_TOK * E_SZ * 4;   // 8 MB
  __bf16* hbf = (__bf16*)(ws + off); off += (long)M_TOK * E_SZ * 2;   // 4 MB (LN out / attn out)
  __bf16* kbf = (__bf16*)(ws + off); off += (long)M_TOK * E_SZ * 2;   // 4 MB
  __bf16* fbf = (__bf16*)(ws + off); off += (long)M_TOK * FF_SZ * 2;  // 16 MB
  __bf16* xb  = (__bf16*)(ws + off); off += (long)M_TOK * E_SZ * 2;   // 4 MB
  __bf16* WkS = (__bf16*)(ws + off); off += (long)L_SZ * E_SZ * E_SZ * 2;
  __bf16* WpS = (__bf16*)(ws + off); off += (long)L_SZ * E_SZ * E_SZ * 2;
  __bf16* W1S = (__bf16*)(ws + off); off += (long)L_SZ * E_SZ * FF_SZ * 2;
  __bf16* W2S = (__bf16*)(ws + off); off += (long)L_SZ * FF_SZ * E_SZ * 2;
  __bf16* lmS = (__bf16*)(ws + off); off += (long)E_SZ * V_SZ * 2;

  // ---- weight swizzle passes (bf16, WMMA fragment order)
  for (int l = 0; l < L_SZ; l++) {
    wswz_kernel<<<dim3(E_SZ / 16, E_SZ / 32), 32, 0, stream>>>(
        Wk + (long)l * H_SZ * E_SZ * HS_SZ, WkS + (long)l * E_SZ * E_SZ, E_SZ, E_SZ, 1);
    wswz_kernel<<<dim3(E_SZ / 16, E_SZ / 32), 32, 0, stream>>>(
        Wproj + (long)l * E_SZ * E_SZ, WpS + (long)l * E_SZ * E_SZ, E_SZ, E_SZ, 0);
    wswz_kernel<<<dim3(FF_SZ / 16, E_SZ / 32), 32, 0, stream>>>(
        W1 + (long)l * E_SZ * FF_SZ, W1S + (long)l * E_SZ * FF_SZ, E_SZ, FF_SZ, 0);
    wswz_kernel<<<dim3(E_SZ / 16, FF_SZ / 32), 32, 0, stream>>>(
        W2 + (long)l * FF_SZ * E_SZ, W2S + (long)l * FF_SZ * E_SZ, FF_SZ, E_SZ, 0);
  }
  wswz_kernel<<<dim3(V_SZ / 16, E_SZ / 32), 32, 0, stream>>>(lm_w, lmS, E_SZ, V_SZ, 0);

  // ---- forward
  embed_kernel<<<M_TOK, 128, 0, stream>>>(inputs, emb, pos, x);

  for (int l = 0; l < L_SZ; l++) {
    ln_kernel<<<M_TOK / 8, 256, 0, stream>>>(x, ln1_g + l * E_SZ, ln1_b + l * E_SZ, hbf);
    gemm_kernel<<<dim3(E_SZ / 64, M_TOK / 256), 256, 0, stream>>>(
        hbf, WkS + (long)l * E_SZ * E_SZ, nullptr, nullptr, kbf,
        M_TOK, E_SZ, E_SZ, GF_OUTBF);
    attn_kernel<<<B_SZ * H_SZ, 64, 0, stream>>>(kbf, hbf);
    gemm_kernel<<<dim3(E_SZ / 64, M_TOK / 256), 256, 0, stream>>>(
        hbf, WpS + (long)l * E_SZ * E_SZ, bproj + l * E_SZ, x, x,
        M_TOK, E_SZ, E_SZ, GF_RES);
    ln_kernel<<<M_TOK / 8, 256, 0, stream>>>(x, ln2_g + l * E_SZ, ln2_b + l * E_SZ, hbf);
    gemm_kernel<<<dim3(FF_SZ / 64, M_TOK / 256), 256, 0, stream>>>(
        hbf, W1S + (long)l * E_SZ * FF_SZ, b1 + l * FF_SZ, nullptr, fbf,
        M_TOK, FF_SZ, E_SZ, GF_RELU | GF_OUTBF);
    gemm_kernel<<<dim3(E_SZ / 64, M_TOK / 256), 256, 0, stream>>>(
        fbf, W2S + (long)l * FF_SZ * E_SZ, b2 + l * E_SZ, x, x,
        M_TOK, E_SZ, FF_SZ, GF_RES);
  }

  cvt_kernel<<<((long)M_TOK * E_SZ + 255) / 256, 256, 0, stream>>>(x, xb, (long)M_TOK * E_SZ);
  gemm_kernel<<<dim3(V_SZ / 64, M_TOK / 256), 256, 0, stream>>>(
      xb, lmS, lm_b, nullptr, logits, M_TOK, V_SZ, E_SZ, 0);

  zero_loss_kernel<<<1, 32, 0, stream>>>(loss);
  loss_kernel<<<M_TOK, 256, 0, stream>>>(logits, targets, loss);
}
